// PainHead_37005438222776
// MI455X (gfx1250) — matrix operand order repaired and checked by reading, not compiled, not run on hardware
//
#include <hip/hip_runtime.h>

typedef __attribute__((ext_vector_type(16))) __bf16 v16bf;
typedef __attribute__((ext_vector_type(8)))  __bf16 v8bf;
typedef __attribute__((ext_vector_type(8)))  float  v8f;
typedef __attribute__((ext_vector_type(8)))  float  v8fv;

#define BM 128
#define BN 128
#define BK 32
#define LDT 40   // padded LDS row (80 B) -> conflict-free, 16B-aligned fragment reads

// ---------- f32 -> bf16 pre-conversion (packed) ----------
__global__ __launch_bounds__(256) void cvt_kernel(
    const float* __restrict__ in, __bf16* __restrict__ out, long n8)
{
    long i = (long)blockIdx.x * 256 + threadIdx.x;
    if (i < n8) {
        v8fv v = *(const v8fv*)(in + i * 8);
        *(v8bf*)(out + i * 8) = __builtin_convertvector(v, v8bf);
    }
}

// out[m][n] = epilogue( sum_k X[m][k] * W[n][k] + bias[n] ),  X/W bf16, f32 accum.
// RELU_BN: val = g*(relu(val) - mu)*rsqrt(var+eps) + beta
// NGUARD : guard n < Nn (only when Nn % 128 != 0). M must be a multiple of 128.
// OUT_BF16: store output rounded to bf16 (for tensors consumed as GEMM A operands).
template<bool RELU_BN, bool NGUARD, bool OUT_BF16>
__global__ __launch_bounds__(256) void gemm_bf16(
    const __bf16* __restrict__ X, const __bf16* __restrict__ W,
    const float* __restrict__ bias,
    const float* __restrict__ gam, const float* __restrict__ bet,
    const float* __restrict__ mu,  const float* __restrict__ var,
    void* __restrict__ outv, int Nn, int K)
{
    __shared__ __bf16 At[BM][LDT];
    __shared__ __bf16 Bt[BN][LDT];

    const int tid  = threadIdx.x;
    const int lane = tid & 31;
    const int wid  = tid >> 5;
    const int lrow = lane & 15;
    const int lhi  = lane >> 4;            // 0 or 1

    const int mbase = blockIdx.x * BM;
    const int nbase = blockIdx.y * BN;
    const int waveM = (wid >> 1) * 32;     // 0,32,64,96
    const int waveN = (wid & 1) * 64;      // 0,64

    // staging coords: chunk of 8 bf16; rows sr and sr+64, cols sc8..sc8+7
    const int sr  = tid >> 2;              // 0..63
    const int sc8 = (tid & 3) << 3;        // 0,8,16,24

    const v8bf vzero = {};
    v8f  acc[2][4] = {};                   // 2 M-tiles x 4 N-tiles of 16x16 f32
    v8bf ar[2], br[2];                     // register double-buffer for next tile

    auto loadFull = [&](int k0) {
        #pragma unroll
        for (int it = 0; it < 2; ++it) {
            int r = sr + it * 64;
            const __bf16* pa = X + (long)(mbase + r) * K + k0 + sc8;
            ar[it] = *(const v8bf*)pa;
            __builtin_prefetch(pa + 2 * BK, 0, 1);           // stream A ahead
            if (NGUARD && (nbase + r) >= Nn) br[it] = vzero;
            else br[it] = *(const v8bf*)(W + (long)(nbase + r) * K + k0 + sc8);
        }
    };

    auto loadTail = [&](int k0) {          // zero-fill out-of-range K (and N)
        #pragma unroll
        for (int it = 0; it < 2; ++it) {
            int r  = sr + it * 64;
            int gk = k0 + sc8;
            v8bf a = vzero, b = vzero;
            const __bf16* pa = X + (long)(mbase + r) * K + gk;
            if (gk + 8 <= K) a = *(const v8bf*)pa;
            else if (gk < K) { for (int e = 0; e < 8; ++e) if (gk + e < K) a[e] = pa[e]; }
            if (!NGUARD || (nbase + r) < Nn) {
                const __bf16* pb = W + (long)(nbase + r) * K + gk;
                if (gk + 8 <= K) b = *(const v8bf*)pb;
                else if (gk < K) { for (int e = 0; e < 8; ++e) if (gk + e < K) b[e] = pb[e]; }
            }
            ar[it] = a; br[it] = b;
        }
    };

    auto ldsStore = [&]() {
        #pragma unroll
        for (int it = 0; it < 2; ++it) {
            int r = sr + it * 64;
            *(v8bf*)&At[r][sc8] = ar[it];
            *(v8bf*)&Bt[r][sc8] = br[it];
        }
    };

    auto compute = [&]() {
        v16bf af[2], bfv[4];
        #pragma unroll
        for (int mt = 0; mt < 2; ++mt) {   // A: ISA 7.12.2 16-bit A layout
            int ard = waveM + mt * 16 + lrow;
            int ak  = lhi * 8;             // lanes 0-15: K 0-7/16-23; 16-31: K 8-15/24-31
            v8bf lo = *(const v8bf*)&At[ard][ak];
            v8bf hi = *(const v8bf*)&At[ard][16 + ak];
            af[mt] = __builtin_shufflevector(lo, hi,
                     0,1,2,3,4,5,6,7,8,9,10,11,12,13,14,15);
        }
        #pragma unroll
        for (int nt = 0; nt < 4; ++nt) {   // B: column-contiguous (B = W^T)
            int brd = waveN + nt * 16 + lrow;
            int bk  = lhi * 8;             // lanes 0-15: K 0-7/16-23; 16-31: K 8-15/24-31
            v8bf lo = *(const v8bf*)&Bt[brd][bk];
            v8bf hi = *(const v8bf*)&Bt[brd][bk + 16];
            bfv[nt] = __builtin_shufflevector(lo, hi,
                      0,1,2,3,4,5,6,7,8,9,10,11,12,13,14,15);
        }
        #pragma unroll
        for (int mt = 0; mt < 2; ++mt)
            #pragma unroll
            for (int nt = 0; nt < 4; ++nt)
                acc[mt][nt] = __builtin_amdgcn_wmma_f32_16x16x32_bf16(
                    false, af[mt], false, bfv[nt], (short)0, acc[mt][nt],
                    false, false);
    };

    const int kfull = K & ~(BK - 1);

    if (kfull > 0) {
        loadFull(0);
        for (int k0 = 0; k0 < kfull; k0 += BK) {
            ldsStore();
            __syncthreads();
            if (k0 + BK < kfull) loadFull(k0 + BK);   // overlap next global load w/ WMMA
            compute();
            __syncthreads();
        }
    }
    if (kfull < K) {                                   // single guarded tail tile
        loadTail(kfull);
        ldsStore();
        __syncthreads();
        compute();
    }

    // ---- epilogue: bias (+ ReLU + BN), optional bf16 store ----
    #pragma unroll
    for (int mt = 0; mt < 2; ++mt) {
        #pragma unroll
        for (int nt = 0; nt < 4; ++nt) {
            int n = nbase + waveN + nt * 16 + lrow;
            if (NGUARD && n >= Nn) continue;
            float bb = bias[n];
            float gg = 1.f, bbe = 0.f, mm = 0.f, rs = 1.f;
            if (RELU_BN) {
                gg  = gam[n];
                bbe = bet[n];
                mm  = mu[n];
                rs  = rsqrtf(var[n] + 1e-5f);
            }
            #pragma unroll
            for (int i = 0; i < 8; ++i) {
                int m = mbase + waveM + mt * 16 + i + lhi * 8;
                float val = acc[mt][nt][i] + bb;
                if (RELU_BN) val = gg * (fmaxf(val, 0.f) - mm) * rs + bbe;
                if (OUT_BF16) ((__bf16*)outv)[(long)m * Nn + n] = (__bf16)val;
                else          ((float*)outv)[(long)m * Nn + n] = val;
            }
        }
    }
}

// Per-row ragged AvgPool1d(k=9, count_include_pad=False); f32 in, bf16 out.
__global__ __launch_bounds__(256) void segpool_kernel(
    const float* __restrict__ h, const int* __restrict__ key,
    __bf16* __restrict__ out, int B, int D)
{
    int i = blockIdx.x;
    int k = key[i];
    int lo = i, hi = i;
    #pragma unroll
    for (int s = 0; s < 4; ++s) if (lo > 0     && key[lo - 1] == k) --lo;
    #pragma unroll
    for (int s = 0; s < 4; ++s) if (hi < B - 1 && key[hi + 1] == k) ++hi;
    float inv = 1.f / (float)(hi - lo + 1);
    for (int d = threadIdx.x; d < D; d += blockDim.x) {
        float s = 0.f;
        for (int j = lo; j <= hi; ++j) s += h[(long)j * D + d];
        out[(long)i * D + d] = (__bf16)(s * inv);
    }
}

// Final H->2 projection + softmax; one wave32 per row.
__global__ __launch_bounds__(256) void head_kernel(
    const float* __restrict__ h2, const float* __restrict__ Wp,
    const float* __restrict__ bp, float* __restrict__ out, int B, int H)
{
    int gwid = (int)((blockIdx.x * blockDim.x + threadIdx.x) >> 5);
    int lane = threadIdx.x & 31;
    if (gwid >= B) return;
    const float* row = h2 + (long)gwid * H;
    float s0 = 0.f, s1 = 0.f;
    for (int k = lane; k < H; k += 32) {
        float x = row[k];
        s0 += x * Wp[k];
        s1 += x * Wp[H + k];
    }
    #pragma unroll
    for (int off = 16; off > 0; off >>= 1) {
        s0 += __shfl_xor(s0, off, 32);
        s1 += __shfl_xor(s1, off, 32);
    }
    if (lane == 0) {
        float o0 = s0 + bp[0], o1 = s1 + bp[1];
        out[2 * gwid + 0] = o0;
        out[2 * gwid + 1] = o1;
        float mx = fmaxf(o0, o1);
        float e0 = __expf(o0 - mx), e1 = __expf(o1 - mx);
        float inv = 1.f / (e0 + e1);
        out[2 * B + 2 * gwid + 0] = e0 * inv;
        out[2 * B + 2 * gwid + 1] = e1 * inv;
    }
}

extern "C" void kernel_launch(void* const* d_in, const int* in_sizes, int n_in,
                              void* d_out, int out_size, void* d_ws, size_t ws_size,
                              hipStream_t stream) {
    (void)in_sizes; (void)n_in; (void)out_size; (void)ws_size;

    const float* img = (const float*)d_in[0];
    const int*   key = (const int*)  d_in[1];
    const float* W3d = (const float*)d_in[2];
    const float* b3d = (const float*)d_in[3];
    const float* W1  = (const float*)d_in[4];
    const float* b1  = (const float*)d_in[5];
    const float* g1  = (const float*)d_in[6];
    const float* be1 = (const float*)d_in[7];
    const float* m1  = (const float*)d_in[8];
    const float* v1  = (const float*)d_in[9];
    const float* W2  = (const float*)d_in[10];
    const float* b2  = (const float*)d_in[11];
    const float* g2  = (const float*)d_in[12];
    const float* be2 = (const float*)d_in[13];
    const float* m2  = (const float*)d_in[14];
    const float* v2  = (const float*)d_in[15];
    const float* Wp  = (const float*)d_in[16];
    const float* bp  = (const float*)d_in[17];

    constexpr int B = 16384, F = 2048, D3 = 600, H = 2048;

    // ---- workspace layout (bf16 tensors first, then f32 h buffer) ----
    __bf16* imgb  = (__bf16*)d_ws;                       // [B,  F ]
    __bf16* W3db  = imgb  + (size_t)B  * F;              // [D3, F ]
    __bf16* W1b   = W3db  + (size_t)D3 * F;              // [H,  D3]
    __bf16* W2b   = W1b   + (size_t)H  * D3;             // [H,  H ]
    __bf16* latb  = W2b   + (size_t)H  * H;              // [B,  D3]
    __bf16* poolb = latb  + (size_t)B  * D3;             // [B,  H ]
    float*  hbuf  = (float*)(poolb + (size_t)B * H);     // [B,  H ] (h, then h2)

    dim3 blk(256);
    auto cvt = [&](const float* src, __bf16* dst, long n) {
        long n8 = n / 8;
        cvt_kernel<<<dim3((unsigned)((n8 + 255) / 256)), blk, 0, stream>>>(src, dst, n8);
    };

    // ---- one-shot f32 -> bf16 conversions ----
    cvt(img, imgb, (long)B * F);
    cvt(W3d, W3db, (long)D3 * F);
    cvt(W1,  W1b,  (long)H * D3);
    cvt(W2,  W2b,  (long)H * H);

    // GEMM1: latent(bf16) = img @ W3d^T + b3d   (N=600 -> N guard; bf16 out)
    gemm_bf16<false, true, true><<<dim3(B / BM, (D3 + BN - 1) / BN), blk, 0, stream>>>(
        imgb, W3db, b3d, nullptr, nullptr, nullptr, nullptr, latb, D3, F);

    // GEMM2: h(f32) = BN(ReLU(latent @ W1^T + b1))   (K=600 handled by tail tile)
    gemm_bf16<true, false, false><<<dim3(B / BM, H / BN), blk, 0, stream>>>(
        latb, W1b, b1, g1, be1, m1, v1, hbuf, H, D3);

    // Ragged segment avg-pool (f32 in, bf16 out)
    segpool_kernel<<<B, blk, 0, stream>>>(hbuf, key, poolb, B, H);

    // GEMM3: h2(f32) = BN(ReLU(pooled @ W2^T + b2))  (overwrites hbuf)
    gemm_bf16<true, false, false><<<dim3(B / BM, H / BN), blk, 0, stream>>>(
        poolb, W2b, b2, g2, be2, m2, v2, hbuf, H, H);

    // logits + softmax
    head_kernel<<<(B * 32 + 255) / 256, blk, 0, stream>>>(
        hbuf, Wp, bp, (float*)d_out, B, H);
}
